// MultiheadAttention_51445118271742
// MI455X (gfx1250) — compile-verified, matrix operand order checked by reading, not run on hardware
//
#include <hip/hip_runtime.h>
#include <hip/hip_bf16.h>

// ---------------- problem constants ----------------
#define S_LEN 2048
#define BATCH 2
#define EMB   1024
#define HEADS 16
#define HDIM  64
#define F3    3072
#define ROWS  (S_LEN * BATCH)                    // 4096
#define QELEMS (BATCH * HEADS * S_LEN * HDIM)    // 4,194,304 per Q/K/V

// ---------------- types ----------------
typedef __attribute__((ext_vector_type(16))) __bf16        v16bf;
typedef __attribute__((ext_vector_type(16))) unsigned short v16u;
typedef __attribute__((ext_vector_type(8)))  unsigned short v8u;
typedef __attribute__((ext_vector_type(8)))  float          v8f;
typedef __attribute__((ext_vector_type(4)))  float          v4f;
typedef unsigned short u16;

// float -> bf16 via native convert (RNE)
__device__ __forceinline__ u16 f2bf(float f) {
    __bf16 h = (__bf16)f;
    return __builtin_bit_cast(u16, h);
}

// Load 16 contiguous bf16 (32B aligned) as a WMMA fragment.
__device__ __forceinline__ v16bf load16bf(const u16* p) {
    return __builtin_bit_cast(v16bf, *(const v16u*)p);
}

__device__ __forceinline__ v8f wmma_bf16(v16bf a, v16bf b, v8f c) {
    return __builtin_amdgcn_wmma_f32_16x16x32_bf16(false, a, false, b, (short)0, c,
                                                   false, false);
}

// =====================================================================
// Kernel 0: elementwise f32 -> bf16 convert (8 elements/thread)
// =====================================================================
__global__ void cvt_bf16_kernel(const float* __restrict__ src,
                                u16* __restrict__ dst, int n) {
    const int base = (blockIdx.x * blockDim.x + threadIdx.x) * 8;
    if (base + 8 > n) return;
    v4f f0 = *(const v4f*)(src + base);
    v4f f1 = *(const v4f*)(src + base + 4);
    v8u o;
#pragma unroll
    for (int i = 0; i < 4; ++i) { o[i] = f2bf(f0[i]); o[4 + i] = f2bf(f1[i]); }
    *(v8u*)(dst + base) = o;
}

// =====================================================================
// Kernel 1: QKV projection (bf16 inputs).  Wave -> 32(M) x 64(N) tile.
//   Q,K stored as [B,H,S,D]; V stored transposed as [B,H,D,S].
// =====================================================================
__global__ void qkv_proj_kernel(const u16* __restrict__ xbf,
                                const u16* __restrict__ wbf,
                                const float* __restrict__ b_qkv,
                                u16* __restrict__ ws) {
    const int tid  = threadIdx.x;
    const int lane = tid & 31;
    const int wv   = tid >> 5;
    const int tile = blockIdx.x * 4 + wv;      // 128 * 48 tiles
    const int mt   = tile / 48;                // 32-row M tile
    const int ng   = tile % 48;                // 64-col N group

    const int koff = (lane >> 4) << 4;
    const u16* xr0 = xbf + (size_t)(mt * 32 + (lane & 15)) * EMB + koff;
    const u16* xr1 = xr0 + 16 * EMB;

    v8f acc[2][4] = {};

    for (int k0 = 0; k0 < EMB; k0 += 32) {
        __builtin_prefetch(xr0 + k0 + 256, 0, 0);
        v16bf a0 = load16bf(xr0 + k0);
        v16bf a1 = load16bf(xr1 + k0);
        const u16* wr = wbf + (size_t)(ng * 64 + (lane & 15)) * EMB + koff + k0;
#pragma unroll
        for (int j = 0; j < 4; ++j) {
            v16bf b = load16bf(wr + (size_t)j * 16 * EMB);
            acc[0][j] = wmma_bf16(a0, b, acc[0][j]);
            acc[1][j] = wmma_bf16(a1, b, acc[1][j]);
        }
    }

#pragma unroll
    for (int j = 0; j < 4; ++j) {
        const int n    = ng * 64 + j * 16 + (lane & 15);   // f in [0,3072)
        const float bv = b_qkv[n];
        const int mat  = n / EMB;                          // 0=Q 1=K 2=V
        const int rem  = n % EMB;
        const int h    = rem / HDIM;
        const int d    = rem % HDIM;
#pragma unroll
        for (int mm = 0; mm < 2; ++mm) {
#pragma unroll
            for (int i = 0; i < 8; ++i) {
                const int m  = mt * 32 + mm * 16 + i + ((lane >> 4) << 3);
                const int s  = m / BATCH;
                const int bb = m % BATCH;
                const u16 bf = f2bf(acc[mm][j][i] + bv);
                size_t idx;
                if (mat < 2) {  // Q / K : [B,H,S,D]
                    idx = (size_t)mat * QELEMS +
                          (((size_t)(bb * HEADS + h) * S_LEN + s) * HDIM + d);
                } else {        // V transposed : [B,H,D,S]
                    idx = (size_t)2 * QELEMS +
                          (((size_t)(bb * HEADS + h) * HDIM + d) * S_LEN + s);
                }
                ws[idx] = bf;
            }
        }
    }
}

// =====================================================================
// Kernel 2: attention.  Block = (b, 16-query-row tile); loops all heads.
// LDS: scores f32 128KB + head-mean f32 128KB + probs bf16 64KB = 320KB.
// =====================================================================
__global__ void attn_kernel(const u16* __restrict__ ws,
                            u16* __restrict__ attnbuf,     // [S,B,E] bf16
                            float* __restrict__ attn_avg)  // [B,S,S] f32
{
    extern __shared__ char smem_raw[];
    float* sc  = (float*)smem_raw;          // [16][2048]
    float* avg = sc + 16 * S_LEN;           // [16][2048]
    u16*   pbf = (u16*)(avg + 16 * S_LEN);  // [16][2048] bf16 probs

    const int tid   = threadIdx.x;          // 256 threads = 8 waves
    const int lane  = tid & 31;
    const int wv    = tid >> 5;
    const int stile = blockIdx.x % (S_LEN / 16);
    const int bb    = blockIdx.x / (S_LEN / 16);

    const u16* Q  = ws;
    const u16* K  = ws + QELEMS;
    const u16* Vt = ws + 2 * (size_t)QELEMS;

    for (int idx = tid; idx < 16 * S_LEN; idx += 256) avg[idx] = 0.0f;
    __syncthreads();

    const int koff = (lane >> 4) << 4;

    for (int h = 0; h < HEADS; ++h) {
        // ---- Q fragments (16 x 64) ----
        const u16* qp = Q + ((size_t)(bb * HEADS + h) * S_LEN +
                             stile * 16 + (lane & 15)) * HDIM + koff;
        const v16bf a0 = load16bf(qp);
        const v16bf a1 = load16bf(qp + 32);

        // ---- scores: wave wv -> keys [wv*256, wv*256+256) ----
#pragma unroll 4
        for (int tt = 0; tt < 16; ++tt) {
            const int t0 = wv * 256 + tt * 16;
            const u16* kp = K + ((size_t)(bb * HEADS + h) * S_LEN +
                                 t0 + (lane & 15)) * HDIM + koff;
            v8f c = {};
            c = wmma_bf16(a0, load16bf(kp),      c);
            c = wmma_bf16(a1, load16bf(kp + 32), c);
            const int n     = lane & 15;
            const int mbase = (lane >> 4) << 3;
#pragma unroll
            for (int i = 0; i < 8; ++i)
                sc[(i + mbase) * S_LEN + t0 + n] = c[i] * 0.125f;
        }
        __syncthreads();

        // ---- softmax: wave wv owns rows 2wv, 2wv+1 ----
        for (int rr = 0; rr < 2; ++rr) {
            const int row = wv * 2 + rr;
            float* rowp = sc  + row * S_LEN;
            float* avgp = avg + row * S_LEN;
            u16*   pp   = pbf + row * S_LEN;
            float mx = -3.402823466e38f;
            for (int j = lane; j < S_LEN; j += 32) mx = fmaxf(mx, rowp[j]);
#pragma unroll
            for (int off = 16; off > 0; off >>= 1)
                mx = fmaxf(mx, __shfl_xor(mx, off, 32));
            float sum = 0.0f;
            for (int j = lane; j < S_LEN; j += 32) {
                float e = __expf(rowp[j] - mx);
                rowp[j] = e;
                sum += e;
            }
#pragma unroll
            for (int off = 16; off > 0; off >>= 1)
                sum += __shfl_xor(sum, off, 32);
            const float inv = 1.0f / sum;
            for (int j = lane; j < S_LEN; j += 32) {
                float p = rowp[j] * inv;
                pp[j] = f2bf(p);                 // bf16 probs for P*V
                avgp[j] += p * (1.0f / HEADS);   // head-mean accumulation
            }
        }
        __syncthreads();

        // ---- P*V : waves 0..3, wave wv -> d-tile [wv*16, +16) ----
        if (wv < 4) {
            v8f o = {};
            const u16* vbase = Vt + ((size_t)(bb * HEADS + h) * HDIM +
                                     wv * 16 + (lane & 15)) * S_LEN + koff;
            const u16* pbase = pbf + (size_t)(lane & 15) * S_LEN + koff;
            for (int t0 = 0; t0 < S_LEN; t0 += 32) {
                o = wmma_bf16(load16bf(pbase + t0), load16bf(vbase + t0), o);
            }
            const int n = lane & 15;
#pragma unroll
            for (int i = 0; i < 8; ++i) {
                const int m = stile * 16 + i + ((lane >> 4) << 3);   // s
                const size_t idx = ((size_t)m * BATCH + bb) * EMB +
                                   h * HDIM + wv * 16 + n;
                attnbuf[idx] = f2bf(o[i]);
            }
        }
        __syncthreads();
    }

    // ---- write head-averaged attention weights: [B,S,S] ----
    for (int idx = tid; idx < 16 * S_LEN; idx += 256) {
        const int rr = idx / S_LEN;
        const int t  = idx % S_LEN;
        const int s  = stile * 16 + rr;
        attn_avg[((size_t)bb * S_LEN + s) * S_LEN + t] = avg[idx];
    }
}

// =====================================================================
// Kernel 3: output projection (bf16 inputs).  Wave -> 32(M) x 64(N).
// =====================================================================
__global__ void out_proj_kernel(const u16* __restrict__ attnbuf,
                                const u16* __restrict__ wbf,
                                const float* __restrict__ b_out,
                                float* __restrict__ out) {
    const int tid  = threadIdx.x;
    const int lane = tid & 31;
    const int wv   = tid >> 5;
    const int tile = blockIdx.x * 4 + wv;      // 128 * 16 tiles
    const int mt   = tile / 16;
    const int ng   = tile % 16;

    const int koff = (lane >> 4) << 4;
    const u16* ar0 = attnbuf + (size_t)(mt * 32 + (lane & 15)) * EMB + koff;
    const u16* ar1 = ar0 + 16 * EMB;

    v8f acc[2][4] = {};

    for (int k0 = 0; k0 < EMB; k0 += 32) {
        v16bf a0 = load16bf(ar0 + k0);
        v16bf a1 = load16bf(ar1 + k0);
        const u16* wr = wbf + (size_t)(ng * 64 + (lane & 15)) * EMB + koff + k0;
#pragma unroll
        for (int j = 0; j < 4; ++j) {
            v16bf b = load16bf(wr + (size_t)j * 16 * EMB);
            acc[0][j] = wmma_bf16(a0, b, acc[0][j]);
            acc[1][j] = wmma_bf16(a1, b, acc[1][j]);
        }
    }

#pragma unroll
    for (int j = 0; j < 4; ++j) {
        const int n    = ng * 64 + j * 16 + (lane & 15);
        const float bv = b_out[n];
#pragma unroll
        for (int mm = 0; mm < 2; ++mm) {
#pragma unroll
            for (int i = 0; i < 8; ++i) {
                const int m = mt * 32 + mm * 16 + i + ((lane >> 4) << 3);
                out[(size_t)m * EMB + n] = acc[mm][j][i] + bv;
            }
        }
    }
}

// =====================================================================
extern "C" void kernel_launch(void* const* d_in, const int* in_sizes, int n_in,
                              void* d_out, int out_size, void* d_ws, size_t ws_size,
                              hipStream_t stream) {
    (void)in_sizes; (void)n_in; (void)out_size; (void)ws_size;

    const float* x     = (const float*)d_in[0];   // [S,B,E]
    const float* w_qkv = (const float*)d_in[1];   // [3E,E]
    const float* b_qkv = (const float*)d_in[2];   // [3E]
    const float* w_out = (const float*)d_in[3];   // [E,E]
    const float* b_out = (const float*)d_in[4];   // [E]

    float* out      = (float*)d_out;                       // [S,B,E]
    float* attn_avg = out + (size_t)S_LEN * BATCH * EMB;   // [B,S,S]

    // workspace layout (u16 elements)
    u16* ws      = (u16*)d_ws;
    u16* attnbuf = ws + 3 * (size_t)QELEMS;                      // [S,B,E]
    u16* xbf     = attnbuf + (size_t)ROWS * EMB;                 // [ROWS,E]
    u16* wqkvbf  = xbf + (size_t)ROWS * EMB;                     // [3E,E]
    u16* woutbf  = wqkvbf + (size_t)F3 * EMB;                    // [E,E]

    // 0) convert activations + weights to bf16 once
    {
        const int nx = ROWS * EMB, nw1 = F3 * EMB, nw2 = EMB * EMB;
        cvt_bf16_kernel<<<nx  / (256 * 8), 256, 0, stream>>>(x,     xbf,    nx);
        cvt_bf16_kernel<<<nw1 / (256 * 8), 256, 0, stream>>>(w_qkv, wqkvbf, nw1);
        cvt_bf16_kernel<<<nw2 / (256 * 8), 256, 0, stream>>>(w_out, woutbf, nw2);
    }

    // 1) QKV projection: 128 M-tiles(32) * 48 N-groups(64), 4 waves/block
    qkv_proj_kernel<<<(128 * 48) / 4, 128, 0, stream>>>(xbf, wqkvbf, b_qkv, ws);

    // 2) attention: one block per (b, 16-row q-tile); 320KB dynamic LDS
    const size_t smem = (size_t)16 * S_LEN * (4 + 4 + 2);
    attn_kernel<<<BATCH * (S_LEN / 16), 256, smem, stream>>>(ws, attnbuf, attn_avg);

    // 3) output projection: 128 M-tiles(32) * 16 N-groups(64), 4 waves/block
    out_proj_kernel<<<(128 * 16) / 4, 128, 0, stream>>>(attnbuf, woutbf, b_out, out);
}
